// MambaBlock_51110110822785
// MI455X (gfx1250) — compile-verified, matrix operand order checked by reading, not run on hardware
//
#include <hip/hip_runtime.h>
#include <hip/hip_bf16.h>
#include <stdint.h>

// ---------------------------------------------------------------------------
// Mamba block for MI455X (gfx1250, wave32).  GEMMs use bf16 WMMA
// (v_wmma_f32_16x16x32_bf16) with f32 accumulation, double-buffered LDS with
// async global->LDS DMA for the A tile (ASYNCcnt path), padded N so there are
// no boundary paths at all.
// ---------------------------------------------------------------------------

#define HID    768
#define INNER  1536
#define NPROJ  (2 * 16 + INNER)   // 1568
#define NPROJ_PAD 1792            // padded to multiple of 256 (GEMM N tile)
#define NSTATE 16
#define KCONV  4
#define SEQ    2048
#define BATCH  2
#define ROWS   (BATCH * SEQ)      // 4096

typedef __attribute__((ext_vector_type(16))) __bf16 v16bf;
typedef __attribute__((ext_vector_type(8)))  float  v8f;

__device__ __forceinline__ unsigned short f2bf(float x) {
    unsigned int u = __float_as_uint(x);
    unsigned int r = u + 0x7FFFu + ((u >> 16) & 1u);   // round-to-nearest-even
    return (unsigned short)(r >> 16);
}
__device__ __forceinline__ float siluf(float x) { return x / (1.0f + __expf(-x)); }

// ---- CDNA5 async global->LDS copy (tracked by ASYNCcnt) --------------------
__device__ __forceinline__ void async_copy_b128(unsigned lds_addr, const void* gaddr) {
    asm volatile("global_load_async_to_lds_b128 %0, %1, off"
                 :: "v"(lds_addr), "v"(gaddr) : "memory");
}
__device__ __forceinline__ void wait_async0() {
    asm volatile("s_wait_asynccnt 0x0" ::: "memory");
}

// ------------------ fp32 -> bf16 convert with column padding ----------------
__global__ void k_cvt_pad(const float* __restrict__ s, unsigned short* __restrict__ d,
                          int rows, int cols, int ldS, int ldD) {
    size_t i = (size_t)blockIdx.x * blockDim.x + threadIdx.x;
    size_t stride = (size_t)gridDim.x * blockDim.x;
    const size_t total = (size_t)rows * ldD;
    for (; i < total; i += stride) {
        const int c = (int)(i % ldD);
        const int r = (int)(i / ldD);
        d[i] = (c < cols) ? f2bf(s[(size_t)r * ldS + c]) : (unsigned short)0;
    }
}

// ------------------------------- layernorm ----------------------------------
__global__ __launch_bounds__(256) void k_layernorm(const float* __restrict__ x,
                                                   const float* __restrict__ w,
                                                   const float* __restrict__ b,
                                                   unsigned short* __restrict__ out) {
    __shared__ float s1[256], s2[256];
    const int row = blockIdx.x;
    const int tid = threadIdx.x;
    const float* xr = x + (size_t)row * HID;
    float s = 0.f, ss = 0.f;
    float v[3];
    for (int i = 0; i < 3; ++i) {
        v[i] = xr[tid + i * 256];
        s += v[i]; ss += v[i] * v[i];
    }
    s1[tid] = s; s2[tid] = ss;
    __syncthreads();
    for (int o = 128; o > 0; o >>= 1) {
        if (tid < o) { s1[tid] += s1[tid + o]; s2[tid] += s2[tid + o]; }
        __syncthreads();
    }
    const float mu  = s1[0] * (1.0f / HID);
    const float var = s2[0] * (1.0f / HID) - mu * mu;
    const float rs  = rsqrtf(var + 1e-5f);
    for (int i = 0; i < 3; ++i) {
        int e = tid + i * 256;
        out[(size_t)row * HID + e] = f2bf((v[i] - mu) * rs * w[e] + b[e]);
    }
}

// ------------------------------- WMMA GEMM ----------------------------------
// C[M,N] (f32) = A[M,K](bf16) * B[K,N](bf16)  (+ residual).
// Requirements: M % 64 == 0, N % 256 == 0 (pad buffers), K % 32 == 0.
// Block tile 64x256, 8 waves in 2(M) x 4(N); each wave a 32x64 sub-tile
// = 2x4 fragments of 16x16, K-step 32.  LDS double-buffered, 1 barrier/iter.
// A tile copied with GLOBAL_LOAD_ASYNC_TO_LDS_B128; B tile register-staged
// (needs K-quad transpose pack) and stored via DS.
#define LDSP 40   // padded row stride (halves); 80B rows keep 16B alignment

// K-pair start for VGPR v / lane half, per CDNA5 ISA 16-bit operand layout
__device__ __forceinline__ int kmap(int v, int khalf) {
    int base = (v < 4) ? (2 * v) : (16 + 2 * (v - 4));
    return base + khalf * 8;
}

union FragH { v16bf v; unsigned int u[8]; };
union AccU  { v8f v; float f[8]; };
union U4    { float4 f; unsigned short h[8]; unsigned int w[4]; };

__global__ __launch_bounds__(256) void k_gemm_bf16(const unsigned short* __restrict__ A,
                                                   const unsigned short* __restrict__ Bw,
                                                   float* __restrict__ outF,
                                                   const float* __restrict__ residual,
                                                   int M, int N, int K) {
    __shared__ unsigned short As[2][64 * LDSP];      // [buf][64 M x 32 K]
    __shared__ unsigned short Bst[2][256 * LDSP];    // [buf][256 N (transposed) x 32 K]

    const int tid  = threadIdx.x;
    const int n0   = blockIdx.x * 256;
    const int m0   = blockIdx.y * 64;

    const int wave   = tid >> 5;
    const int lane   = tid & 31;
    const int lane16 = lane & 15;
    const int khalf  = lane >> 4;
    const int wmBase = (wave & 1) * 32;
    const int wnBase = (wave >> 1) * 64;

    // global-load mapping: A: 64 rows x 4 groups of 8 halves (16B per thread)
    const int arow = tid >> 2, acg = tid & 3;
    // B: 8 k-quads (4 rows each) x 32 n-groups of 8 halves
    const int bkp = tid >> 5, bng = tid & 31;

    const unsigned short* aP = A + (size_t)(m0 + arow) * K + acg * 8;
    const unsigned short* bP = Bw + (size_t)(4 * bkp) * N + n0 + bng * 8;

    const unsigned ldsA0 = (unsigned)(uintptr_t)&As[0][arow * LDSP + acg * 8];
    const unsigned ldsA1 = (unsigned)(uintptr_t)&As[1][arow * LDSP + acg * 8];

    // ---- prologue: async DMA first A slab into LDS[0]; B slab into regs ----
    async_copy_b128(ldsA0, aP);
    U4 gb0, gb1, gb2, gb3;
    gb0.f = *(const float4*)(bP);
    gb1.f = *(const float4*)(bP + (size_t)N);
    gb2.f = *(const float4*)(bP + (size_t)2 * N);
    gb3.f = *(const float4*)(bP + (size_t)3 * N);

    v8f acc[2][4];
    const v8f vzero = {0.f, 0.f, 0.f, 0.f, 0.f, 0.f, 0.f, 0.f};
    for (int i = 0; i < 2; ++i)
        for (int j = 0; j < 4; ++j) acc[i][j] = vzero;

    int buf = 0;
    for (int k0 = 0; k0 < K; k0 += 32) {
        // ---- stage B registers (transposed) into LDS[buf] ----
        for (int i = 0; i < 8; ++i) {
            uint2 p;
            p.x = (unsigned int)gb0.h[i] | ((unsigned int)gb1.h[i] << 16);
            p.y = (unsigned int)gb2.h[i] | ((unsigned int)gb3.h[i] << 16);
            *(uint2*)(&Bst[buf][(bng * 8 + i) * LDSP + 4 * bkp]) = p;
        }
        // ---- issue next B slab global loads (overlap with compute) ----
        aP += 32;
        bP += (size_t)32 * N;
        const bool more = (k0 + 32 < K);
        if (more) {
            gb0.f = *(const float4*)(bP);
            gb1.f = *(const float4*)(bP + (size_t)N);
            gb2.f = *(const float4*)(bP + (size_t)2 * N);
            gb3.f = *(const float4*)(bP + (size_t)3 * N);
        }
        // this wave's async A writes into LDS[buf] must be visible pre-barrier
        wait_async0();
        __syncthreads();
        // safe to DMA the next A slab into LDS[buf^1] now (all waves finished
        // reading it for iteration k0-32 before they signaled the barrier)
        if (more)
            async_copy_b128(buf ? ldsA0 : ldsA1, aP);

        // ---- fragments + 8 WMMA ----
        FragH fa[2], fb[4];
        for (int ms = 0; ms < 2; ++ms) {
            const int m = wmBase + ms * 16 + lane16;
            for (int v = 0; v < 8; ++v)
                fa[ms].u[v] = *(const unsigned int*)(&As[buf][m * LDSP + kmap(v, khalf)]);
        }
        for (int ns = 0; ns < 4; ++ns) {
            const int n = wnBase + ns * 16 + lane16;
            for (int v = 0; v < 8; ++v)
                fb[ns].u[v] = *(const unsigned int*)(&Bst[buf][n * LDSP + kmap(v, khalf)]);
        }
        for (int ms = 0; ms < 2; ++ms)
            for (int ns = 0; ns < 4; ++ns)
                acc[ms][ns] = __builtin_amdgcn_wmma_f32_16x16x32_bf16(
                    false, fa[ms].v, false, fb[ns].v, (short)0, acc[ms][ns], false, false);
        buf ^= 1;
    }

    // ---- epilogue: C layout -> lane16 = N col, VGPR r -> M = r + 8*khalf ----
    if (residual) {
        for (int ms = 0; ms < 2; ++ms)
            for (int ns = 0; ns < 4; ++ns) {
                AccU au; au.v = acc[ms][ns];
                const int gn = n0 + wnBase + ns * 16 + lane16;
                for (int r = 0; r < 8; ++r) {
                    const int gm = m0 + wmBase + ms * 16 + 8 * khalf + r;
                    const size_t off = (size_t)gm * N + gn;
                    outF[off] = au.f[r] + residual[off];
                }
            }
    } else {
        for (int ms = 0; ms < 2; ++ms)
            for (int ns = 0; ns < 4; ++ns) {
                AccU au; au.v = acc[ms][ns];
                const int gn = n0 + wnBase + ns * 16 + lane16;
                for (int r = 0; r < 8; ++r) {
                    const int gm = m0 + wmBase + ms * 16 + 8 * khalf + r;
                    outF[(size_t)gm * N + gn] = au.f[r];
                }
            }
    }
}

// -------------------- depthwise causal conv (K=4) + SiLU --------------------
__global__ void k_conv_silu(const float* __restrict__ xp,       // [ROWS][2*INNER], u = cols 0..INNER-1
                            const float* __restrict__ cw,       // [INNER][1][KCONV]
                            float* __restrict__ ucF,
                            unsigned short* __restrict__ ucH) {
    size_t idx = (size_t)blockIdx.x * blockDim.x + threadIdx.x;
    const size_t total = (size_t)ROWS * INNER;
    if (idx >= total) return;
    const int ch = (int)(idx % INNER);
    const int s  = (int)((idx / INNER) % SEQ);
    const int b  = (int)(idx / ((size_t)INNER * SEQ));
    float acc = 0.f;
    for (int j = 0; j < KCONV; ++j) {
        const int t = s - (KCONV - 1) + j;
        if (t >= 0)
            acc += cw[ch * KCONV + j] * xp[((size_t)b * SEQ + t) * (2 * INNER) + ch];
    }
    const float v = siluf(acc);
    ucF[idx] = v;
    ucH[idx] = f2bf(v);
}

// ----------------------------- selective scan -------------------------------
// grid: (INNER/16, BATCH), block 256 -> 16 lane-groups of 16 states each.
__global__ __launch_bounds__(256) void k_scan(const float* __restrict__ proj, // [B][S][ldP]
                                              const float* __restrict__ uc,   // [B][S][INNER]
                                              const float* __restrict__ A_log,
                                              const float* __restrict__ Dv,
                                              unsigned short* __restrict__ yH,
                                              int ldP) {
    const int b   = blockIdx.y;
    const int ch  = blockIdx.x * 16 + (threadIdx.x >> 4);
    const int n   = threadIdx.x & 15;
    const float An  = -__expf(A_log[n]);
    const float Dch = Dv[ch];
    const float* pb = proj + (size_t)b * SEQ * ldP;
    const float* ub = uc   + (size_t)b * SEQ * INNER;
    float h = 0.f;
    for (int t = 0; t < SEQ; ++t) {
        const float* pt = pb + (size_t)t * ldP;
        const float Bv   = pt[n];
        const float Cv   = pt[NSTATE + n];
        const float draw = pt[2 * NSTATE + ch];
        const float u    = ub[(size_t)t * INNER + ch];
        const float delta = (draw > 20.f) ? draw : log1pf(__expf(draw));
        h = __expf(delta * An) * h + (delta * u) * Bv;
        float c = h * Cv;
        c += __shfl_xor(c, 8, 16);
        c += __shfl_xor(c, 4, 16);
        c += __shfl_xor(c, 2, 16);
        c += __shfl_xor(c, 1, 16);
        if (n == 0)
            yH[((size_t)b * SEQ + t) * INNER + ch] = f2bf(c + Dch * u);
    }
}

// ------------------------------ gate multiply -------------------------------
__global__ void k_gate_mul(const float* __restrict__ yo,   // [ROWS][INNER]
                           const float* __restrict__ xp,   // [ROWS][2*INNER], gate = cols INNER..
                           unsigned short* __restrict__ zH) {
    size_t idx = (size_t)blockIdx.x * blockDim.x + threadIdx.x;
    const size_t total = (size_t)ROWS * INNER;
    if (idx >= total) return;
    const size_t row = idx / INNER;
    const int    ch  = (int)(idx % INNER);
    const float g = xp[row * (2 * INNER) + INNER + ch];
    zH[idx] = f2bf(yo[idx] * siluf(g));
}

// ---------------------------------------------------------------------------
extern "C" void kernel_launch(void* const* d_in, const int* in_sizes, int n_in,
                              void* d_out, int out_size, void* d_ws, size_t ws_size,
                              hipStream_t stream) {
    (void)in_sizes; (void)n_in; (void)out_size; (void)ws_size;
    const float* x        = (const float*)d_in[0];
    const float* norm_w   = (const float*)d_in[1];
    const float* norm_b   = (const float*)d_in[2];
    const float* W_in     = (const float*)d_in[3];
    const float* conv_w   = (const float*)d_in[4];
    const float* W_xproj  = (const float*)d_in[5];
    const float* A_log    = (const float*)d_in[6];
    const float* Dv       = (const float*)d_in[7];
    const float* W_ssm    = (const float*)d_in[8];
    const float* W_out    = (const float*)d_in[9];
    float* out = (float*)d_out;

    // ---- workspace layout ----
    size_t off = 0;
    auto alloc = [&](size_t bytes) -> char* {
        char* p = (char*)d_ws + off;
        off = (off + bytes + 255) & ~(size_t)255;
        return p;
    };
    unsigned short* xnH   = (unsigned short*)alloc((size_t)ROWS * HID * 2);
    unsigned short* WinH  = (unsigned short*)alloc((size_t)HID * 2 * INNER * 2);
    unsigned short* WxpH  = (unsigned short*)alloc((size_t)INNER * NPROJ_PAD * 2);
    unsigned short* WsoH  = (unsigned short*)alloc((size_t)INNER * INNER * 2);
    unsigned short* WoH   = (unsigned short*)alloc((size_t)INNER * HID * 2);
    float*          xpF   = (float*)alloc((size_t)ROWS * 2 * INNER * 4);
    float*          ucF   = (float*)alloc((size_t)ROWS * INNER * 4);
    unsigned short* ucH   = (unsigned short*)alloc((size_t)ROWS * INNER * 2);
    float*          projF = (float*)alloc((size_t)ROWS * NPROJ_PAD * 4);
    unsigned short* yH    = (unsigned short*)alloc((size_t)ROWS * INNER * 2);
    float*          yoF   = (float*)alloc((size_t)ROWS * INNER * 4);
    unsigned short* zH    = (unsigned short*)alloc((size_t)ROWS * INNER * 2);

    // ---- 1. weight conversion fp32 -> bf16 (W_xproj padded 1568->1792) ----
    k_cvt_pad<<<512, 256, 0, stream>>>(W_in,    WinH, HID,   2 * INNER, 2 * INNER, 2 * INNER);
    k_cvt_pad<<<512, 256, 0, stream>>>(W_xproj, WxpH, INNER, NPROJ,     NPROJ,     NPROJ_PAD);
    k_cvt_pad<<<512, 256, 0, stream>>>(W_ssm,   WsoH, INNER, INNER,     INNER,     INNER);
    k_cvt_pad<<<512, 256, 0, stream>>>(W_out,   WoH,  INNER, HID,       HID,       HID);

    // ---- 2. layernorm -> bf16 ----
    k_layernorm<<<ROWS, 256, 0, stream>>>(x, norm_w, norm_b, xnH);

    // ---- 3. xp = xn @ W_in  (4096 x 768 x 3072) ----
    k_gemm_bf16<<<dim3((2 * INNER) / 256, ROWS / 64), 256, 0, stream>>>(
        xnH, WinH, xpF, nullptr, ROWS, 2 * INNER, HID);

    // ---- 4. depthwise conv + SiLU ----
    k_conv_silu<<<(unsigned)(((size_t)ROWS * INNER + 255) / 256), 256, 0, stream>>>(
        xpF, conv_w, ucF, ucH);

    // ---- 5. proj = uc @ W_xproj  (4096 x 1536 x 1792(pad)) ----
    k_gemm_bf16<<<dim3(NPROJ_PAD / 256, ROWS / 64), 256, 0, stream>>>(
        ucH, WxpH, projF, nullptr, ROWS, NPROJ_PAD, INNER);

    // ---- 6. selective scan -> y (bf16) ----
    k_scan<<<dim3(INNER / 16, BATCH), 256, 0, stream>>>(projF, ucF, A_log, Dv, yH, NPROJ_PAD);

    // ---- 7. yo = y @ W_ssm_out  (4096 x 1536 x 1536) ----
    k_gemm_bf16<<<dim3(INNER / 256, ROWS / 64), 256, 0, stream>>>(
        yH, WsoH, yoF, nullptr, ROWS, INNER, INNER);

    // ---- 8. z = yo * silu(gate) ----
    k_gate_mul<<<(unsigned)(((size_t)ROWS * INNER + 255) / 256), 256, 0, stream>>>(
        yoF, xpF, zH);

    // ---- 9. out = z @ W_out + residual  (4096 x 1536 x 768) ----
    k_gemm_bf16<<<dim3(HID / 256, ROWS / 64), 256, 0, stream>>>(
        zH, WoH, out, x, ROWS, HID, INNER);
}
